// MultiHeadAttentionQuantum_65481071397461
// MI455X (gfx1250) — compile-verified, matrix operand order checked by reading, not run on hardware
//
#include <hip/hip_runtime.h>
#include <hip/hip_bf16.h>
#include <math.h>

#define EMBED 1024
#define BATCH 4
#define SEQ   2048
#define HEADS 16
#define HDIM  64
#define ROWS  (BATCH * SEQ)   // 8192

typedef __attribute__((ext_vector_type(16))) __bf16 v16bf;
typedef __attribute__((ext_vector_type(8)))  __bf16 v8bf;
typedef __attribute__((ext_vector_type(2)))  __bf16 v2bf;
typedef __attribute__((ext_vector_type(8)))  float  v8f;

__device__ __forceinline__ v16bf cat8(v8bf a, v8bf b) {
  return __builtin_shufflevector(a, b, 0, 1, 2, 3, 4, 5, 6, 7,
                                 8, 9, 10, 11, 12, 13, 14, 15);
}

__device__ __forceinline__ v8bf cvt8(float4 f0, float4 f1) {
  v8bf r;
  r[0] = (__bf16)f0.x; r[1] = (__bf16)f0.y; r[2] = (__bf16)f0.z; r[3] = (__bf16)f0.w;
  r[4] = (__bf16)f1.x; r[5] = (__bf16)f1.y; r[6] = (__bf16)f1.z; r[7] = (__bf16)f1.w;
  return r;
}

// ---------------------------------------------------------------------------
// GEMM:  C[ROWS, EMBED] = A[ROWS, EMBED] @ W[EMBED, EMBED] + bias
// fp32 in/out, bf16 WMMA, f32 accumulate.
// Block = 256 threads (8 waves) -> 256x64 output tile; wave = 32x64 strip.
// All four B fragments are preloaded before the 8 WMMAs so ds waits stagger
// and WMMAs issue back-to-back.
// Grid = (ROWS/256, EMBED/64).
// ---------------------------------------------------------------------------
__global__ void __launch_bounds__(256)
gemm_bias_bf16(const float* __restrict__ A, const float* __restrict__ W,
               const float* __restrict__ bias, float* __restrict__ C) {
  const int tid  = threadIdx.x;
  const int wave = tid >> 5;
  const int lane = tid & 31;
  const int hi = lane >> 4;      // K-half selector
  const int lo = lane & 15;      // row (A) / column (B,C) within tile
  const int m0 = (blockIdx.x * 8 + wave) * 32;
  const int n0 = blockIdx.y * 64;

  // transposed bf16 W tile: Wt[n][k], n in [0,64), k in [0,32)
  __shared__ __align__(16) __bf16 Wt[64 * 32];

  // staging: thread -> (k-pair, 4-column group): 16 pairs x 16 groups = 256
  const int sk2 = (tid & 15) * 2;        // even k within tile
  const int snn = (tid >> 4) * 4;        // n within tile, group of 4

  const float4* arowU = (const float4*)(A + (size_t)(m0 + lo) * EMBED);
  const float4* arowL = (const float4*)(A + (size_t)(m0 + 16 + lo) * EMBED);

  v8f acc[2][4] = {};

  for (int k0 = 0; k0 < EMBED; k0 += 32) {
    // ---- load W[k0+sk2..+1][n0+snn..+3] (two coalesced b128) ----
    const float4 wa = *(const float4*)(W + (size_t)(k0 + sk2) * EMBED + n0 + snn);
    const float4 wb = *(const float4*)(W + (size_t)(k0 + sk2 + 1) * EMBED + n0 + snn);
    __syncthreads();                       // previous iteration's reads done
    {
      v2bf p;
      p[0] = (__bf16)wa.x; p[1] = (__bf16)wb.x; *(v2bf*)(&Wt[(snn + 0) * 32 + sk2]) = p;
      p[0] = (__bf16)wa.y; p[1] = (__bf16)wb.y; *(v2bf*)(&Wt[(snn + 1) * 32 + sk2]) = p;
      p[0] = (__bf16)wa.z; p[1] = (__bf16)wb.z; *(v2bf*)(&Wt[(snn + 2) * 32 + sk2]) = p;
      p[0] = (__bf16)wa.w; p[1] = (__bf16)wb.w; *(v2bf*)(&Wt[(snn + 3) * 32 + sk2]) = p;
    }

    // ---- two A fragments (rows m0+lo, m0+16+lo): b128 loads ----
    const int fo = (k0 >> 2) + hi * 2;     // float4 index of k0 + hi*8
    const v16bf aU = cat8(cvt8(arowU[fo], arowU[fo + 1]),
                          cvt8(arowU[fo + 4], arowU[fo + 5]));
    const v16bf aL = cat8(cvt8(arowL[fo], arowL[fo + 1]),
                          cvt8(arowL[fo + 4], arowL[fo + 5]));
    if (k0 + 32 < EMBED) {                 // global_prefetch_b8, near scope
      __builtin_prefetch(A + (size_t)(m0 + lo) * EMBED + k0 + 32 + hi * 8, 0, 3);
      __builtin_prefetch(A + (size_t)(m0 + 16 + lo) * EMBED + k0 + 32 + hi * 8, 0, 3);
    }

    __syncthreads();                       // Wt visible to all waves

    // ---- preload all four B fragments (8 ds_load_b128, staggered waits) ----
    v16bf bfr[4];
#pragma unroll
    for (int t = 0; t < 4; ++t)
      bfr[t] = cat8(*(const v8bf*)(&Wt[(t * 16 + lo) * 32 + hi * 8]),
                    *(const v8bf*)(&Wt[(t * 16 + lo) * 32 + 16 + hi * 8]));

#pragma unroll
    for (int t = 0; t < 4; ++t) {
      acc[0][t] = __builtin_amdgcn_wmma_f32_16x16x32_bf16(
          false, aU, false, bfr[t], (short)0, acc[0][t], false, false);
      acc[1][t] = __builtin_amdgcn_wmma_f32_16x16x32_bf16(
          false, aL, false, bfr[t], (short)0, acc[1][t], false, false);
    }
  }

#pragma unroll
  for (int half = 0; half < 2; ++half) {
#pragma unroll
    for (int t = 0; t < 4; ++t) {
      const int n = n0 + t * 16 + lo;
      const float bv = bias[n];
#pragma unroll
      for (int r = 0; r < 8; ++r) {
        const int m = m0 + half * 16 + r + 8 * hi;   // C layout: row = r + 8*hi
        C[(size_t)m * EMBED + n] = acc[half][t][r] + bv;
      }
    }
  }
}

// ---------------------------------------------------------------------------
// quantum_layer: in-place cumprod(cos(x)) along the 1024-feature axis.
// One wave per row; lane owns a contiguous 32-element chunk (8 x float4);
// wave32 scan of per-lane chunk products via __shfl_up.
// ---------------------------------------------------------------------------
__global__ void __launch_bounds__(256)
quantum_cumprod(float* __restrict__ data) {
  const int wave = threadIdx.x >> 5;
  const int lane = threadIdx.x & 31;
  const int row  = blockIdx.x * 8 + wave;

  float4* p = (float4*)(data + (size_t)row * EMBED + lane * 32);
  float4 v[8];
#pragma unroll
  for (int i = 0; i < 8; ++i) v[i] = p[i];

  float run = 1.0f;
#pragma unroll
  for (int i = 0; i < 8; ++i) {
    run *= cosf(v[i].x); v[i].x = run;
    run *= cosf(v[i].y); v[i].y = run;
    run *= cosf(v[i].z); v[i].z = run;
    run *= cosf(v[i].w); v[i].w = run;
  }

  float incl = run;
#pragma unroll
  for (int off = 1; off < 32; off <<= 1) {
    float s = __shfl_up(incl, off, 32);
    incl = (lane >= off) ? incl * s : incl;
  }
  float pref = __shfl_up(incl, 1, 32);
  if (lane == 0) pref = 1.0f;

#pragma unroll
  for (int i = 0; i < 8; ++i) {
    v[i].x *= pref; v[i].y *= pref; v[i].z *= pref; v[i].w *= pref;
    p[i] = v[i];
  }
}

// ---------------------------------------------------------------------------
// Flash attention: block = 128 threads (4 waves), all on one (b,h).
// Wave w owns q-tile (blockIdx.y*4 + w)*16.  Per 32-kv chunk, the block
// cooperatively stages into LDS:
//   Kl[kv][d]  (row-major bf16)   -> S  B-fragments = 2 x ds_load_b128
//   Vt[d][kv]  (transposed bf16)  -> PV B-fragments = 2 x ds_load_b128
// All fragments of a pass are preloaded before the WMMA burst.
// Online softmax in f32; P staged per-wave through LDS (C->A layout).
// scale = 1/sqrt(64) = 0.125.   Grid = (B*H, SEQ/64).
// ---------------------------------------------------------------------------
__global__ void __launch_bounds__(128)
flash_attn(const float* __restrict__ Q, const float* __restrict__ K,
           const float* __restrict__ V, float* __restrict__ O) {
  const int bh = blockIdx.x;          // 0..63
  const int b  = bh / HEADS;
  const int h  = bh % HEADS;
  const int tid  = threadIdx.x;
  const int wave = tid >> 5;
  const int lane = tid & 31;
  const int hi = lane >> 4;
  const int lo = lane & 15;
  const int q0 = (blockIdx.y * 4 + wave) * 16;
  const size_t bt0  = (size_t)b * SEQ;
  const size_t hoff = (size_t)h * HDIM;

  __shared__ __align__(16) __bf16 Kl[32 * 64];       // K chunk, row-major
  __shared__ __align__(16) __bf16 Vt[64 * 32];       // V chunk, transposed
  __shared__ __align__(16) __bf16 plds[4 * 16 * 32]; // per-wave P staging
  __bf16* pw = &plds[wave * 512];

  // staging maps
  const int skv = tid >> 2;            // 0..31  (kv row for K staging)
  const int sdq = (tid & 3) * 16;      // 0,16,32,48 (d offset for K staging)
  const int vkv = (tid & 15) * 2;      // even kv (pair) for V staging
  const int vdg = (tid >> 4) * 8;      // d group of 8 for V staging

  // Q fragments: 16x64 split into two 16x32 A-fragments (b128 loads, once)
  v16bf aQ[2];
#pragma unroll
  for (int d = 0; d < 2; ++d) {
    const float4* qp = (const float4*)(Q + (bt0 + q0 + lo) * EMBED + hoff + d * 32);
    aQ[d] = cat8(cvt8(qp[hi * 2], qp[hi * 2 + 1]),
                 cvt8(qp[hi * 2 + 4], qp[hi * 2 + 5]));
  }

  v8f o[4] = {};                       // 16x64 output accumulator (4 C-tiles)
  float rmax[8], rsum[8];
#pragma unroll
  for (int r = 0; r < 8; ++r) { rmax[r] = -INFINITY; rsum[r] = 0.0f; }

  for (int kv0 = 0; kv0 < SEQ; kv0 += 32) {
    // ---- cooperative staging of K chunk (row-major) and V chunk (transposed)
    const float4* kp = (const float4*)(K + (bt0 + kv0 + skv) * EMBED + hoff + sdq);
    const float4 kf0 = kp[0], kf1 = kp[1], kf2 = kp[2], kf3 = kp[3];
    const float4* vpa = (const float4*)(V + (bt0 + kv0 + vkv) * EMBED + hoff + vdg);
    const float4* vpb = (const float4*)(V + (bt0 + kv0 + vkv + 1) * EMBED + hoff + vdg);
    const float4 va0 = vpa[0], va1 = vpa[1];
    const float4 vb0 = vpb[0], vb1 = vpb[1];
    if (kv0 + 32 < SEQ)
      __builtin_prefetch(K + (bt0 + kv0 + 32 + skv) * EMBED + hoff + sdq, 0, 3);

    __syncthreads();                     // previous chunk's LDS reads done
    *(v8bf*)(&Kl[skv * 64 + sdq])     = cvt8(kf0, kf1);
    *(v8bf*)(&Kl[skv * 64 + sdq + 8]) = cvt8(kf2, kf3);
    {
      v2bf p;
      const float fa[8] = {va0.x, va0.y, va0.z, va0.w, va1.x, va1.y, va1.z, va1.w};
      const float fb[8] = {vb0.x, vb0.y, vb0.z, vb0.w, vb1.x, vb1.y, vb1.z, vb1.w};
#pragma unroll
      for (int i = 0; i < 8; ++i) {
        p[0] = (__bf16)fa[i]; p[1] = (__bf16)fb[i];
        *(v2bf*)(&Vt[(vdg + i) * 32 + vkv]) = p;
      }
    }
    __syncthreads();                     // staged tiles visible

    // ---- S = Q @ K^T: preload all 4 K fragments, then 4 WMMAs ----
    v16bf bK[2][2];
#pragma unroll
    for (int j = 0; j < 2; ++j)
#pragma unroll
      for (int d = 0; d < 2; ++d)
        bK[j][d] = cat8(
            *(const v8bf*)(&Kl[(j * 16 + lo) * 64 + d * 32 + hi * 8]),
            *(const v8bf*)(&Kl[(j * 16 + lo) * 64 + d * 32 + 16 + hi * 8]));
    v8f s[2];
#pragma unroll
    for (int j = 0; j < 2; ++j) {
      v8f acc = {};
      acc = __builtin_amdgcn_wmma_f32_16x16x32_bf16(
          false, aQ[0], false, bK[j][0], (short)0, acc, false, false);
      acc = __builtin_amdgcn_wmma_f32_16x16x32_bf16(
          false, aQ[1], false, bK[j][1], (short)0, acc, false, false);
      s[j] = acc;
    }

    // ---- online softmax over the 32 new columns (row = r + 8*hi) ----
#pragma unroll
    for (int r = 0; r < 8; ++r) {
      float s0 = s[0][r] * 0.125f;
      float s1 = s[1][r] * 0.125f;
      float m = fmaxf(s0, s1);
#pragma unroll
      for (int off = 1; off < 16; off <<= 1)    // stays within 16-lane half
        m = fmaxf(m, __shfl_xor(m, off, 32));
      const float nm   = fmaxf(rmax[r], m);
      const float corr = __expf(rmax[r] - nm);
      rmax[r] = nm;
      const float p0 = __expf(s0 - nm);
      const float p1 = __expf(s1 - nm);
      float ps = p0 + p1;
#pragma unroll
      for (int off = 1; off < 16; off <<= 1)
        ps += __shfl_xor(ps, off, 32);
      rsum[r] = rsum[r] * corr + ps;
#pragma unroll
      for (int t = 0; t < 4; ++t) o[t][r] *= corr;
      pw[(r + 8 * hi) * 32 + lo]      = (__bf16)p0;
      pw[(r + 8 * hi) * 32 + 16 + lo] = (__bf16)p1;
    }
    __syncthreads();                     // P visible (and Vt reads ordered)

    // ---- preload P and all 4 V fragments, then 4 WMMAs ----
    {
      const v16bf aP = cat8(*(const v8bf*)(&pw[lo * 32 + hi * 8]),
                            *(const v8bf*)(&pw[lo * 32 + 16 + hi * 8]));
      v16bf bV[4];
#pragma unroll
      for (int t = 0; t < 4; ++t)
        bV[t] = cat8(
            *(const v8bf*)(&Vt[(t * 16 + lo) * 32 + hi * 8]),
            *(const v8bf*)(&Vt[(t * 16 + lo) * 32 + 16 + hi * 8]));
#pragma unroll
      for (int t = 0; t < 4; ++t)
        o[t] = __builtin_amdgcn_wmma_f32_16x16x32_bf16(
            false, aP, false, bV[t], (short)0, o[t], false, false);
    }
  }

  // ---- normalize and write combined-heads layout (B*T, EMBED) ----
#pragma unroll
  for (int r = 0; r < 8; ++r) {
    const float inv = 1.0f / rsum[r];
    const size_t row = bt0 + q0 + r + 8 * hi;
#pragma unroll
    for (int t = 0; t < 4; ++t)
      O[row * EMBED + hoff + t * 16 + lo] = o[t][r] * inv;
  }
}

// ---------------------------------------------------------------------------
extern "C" void kernel_launch(void* const* d_in, const int* in_sizes, int n_in,
                              void* d_out, int out_size, void* d_ws, size_t ws_size,
                              hipStream_t stream) {
  const float* x  = (const float*)d_in[0];
  const float* Wq = (const float*)d_in[1];
  const float* bq = (const float*)d_in[2];
  const float* Wk = (const float*)d_in[3];
  const float* bk = (const float*)d_in[4];
  const float* Wv = (const float*)d_in[5];
  const float* bv = (const float*)d_in[6];
  const float* Wo = (const float*)d_in[7];
  const float* bo = (const float*)d_in[8];
  float* out = (float*)d_out;

  float* ws = (float*)d_ws;
  const size_t MN = (size_t)ROWS * EMBED;      // 8M floats = 32 MB each
  float* qb = ws;                              // quantum(x@Wq+bq)  -> KEY role
  float* kb = ws + MN;                         // quantum(x@Wk+bk)  -> QUERY role
  float* vb = ws + 2 * MN;                     // x@Wv+bv
  float* ab = ws + 3 * MN;                     // attention output (combined heads)

  const dim3 gg(ROWS / 256, EMBED / 64);       // (32, 16)
  gemm_bias_bf16<<<gg, 256, 0, stream>>>(x, Wq, bq, qb);
  gemm_bias_bf16<<<gg, 256, 0, stream>>>(x, Wk, bk, kb);
  gemm_bias_bf16<<<gg, 256, 0, stream>>>(x, Wv, bv, vb);

  quantum_cumprod<<<ROWS / 8, 256, 0, stream>>>(qb);
  quantum_cumprod<<<ROWS / 8, 256, 0, stream>>>(kb);

  // Reference swaps roles: query = heads(quantum(k-proj)), key = heads(quantum(q-proj))
  flash_attn<<<dim3(BATCH * HEADS, SEQ / 64), 128, 0, stream>>>(kb, qb, vb, ab);

  gemm_bias_bf16<<<gg, 256, 0, stream>>>(ab, Wo, bo, out);
}
